// SelfMultiheadAttention_23029614641616
// MI455X (gfx1250) — matrix operand order repaired and checked by reading, not compiled, not run on hardware
//
#include <hip/hip_runtime.h>
#include <hip/hip_bf16.h>

// ---------------- problem constants ----------------
#define LSEQ 2048
#define EDIM 1024
#define NH   16
#define HD   64
#define SCALE_Q 0.125f   // D^-0.5 = 64^-0.5

typedef __attribute__((ext_vector_type(16))) __bf16          v16bf;
typedef __attribute__((ext_vector_type(8)))  float           v8f;
typedef __attribute__((ext_vector_type(8)))  unsigned short  v8us;
typedef __attribute__((ext_vector_type(16))) unsigned short  v16us;
typedef __attribute__((ext_vector_type(8)))  unsigned char   v8uc;

union V16Cast { v16us u; v16bf b; };

static __device__ __forceinline__ v16bf us2bf(v16us u) { V16Cast t; t.u = u; return t.b; }

static __device__ __forceinline__ unsigned short f2bfbits(float f) {
  unsigned u; __builtin_memcpy(&u, &f, 4);
  unsigned r = u + 0x7FFFu + ((u >> 16) & 1u);
  return (unsigned short)(r >> 16);
}
static __device__ __forceinline__ float bf2f(unsigned short h) {
  unsigned u = ((unsigned)h) << 16;
  float f; __builtin_memcpy(&f, &u, 4);
  return f;
}

// A fragment: 16x32 bf16, row-major source, `row` is the absolute row for this lane.
static __device__ __forceinline__ v16bf load_a_frag(const unsigned short* __restrict__ base,
                                                    int ld, int row, int k0, int lane) {
  int hf = lane >> 4;
  const unsigned short* p = base + (size_t)row * ld + k0 + (hf << 3);
  v8us a0 = *(const v8us*)p;
  v8us a1 = *(const v8us*)(p + 16);
  v16us a;
#pragma unroll
  for (int i = 0; i < 8; ++i) { a[i] = a0[i]; a[i + 8] = a1[i]; }
  return us2bf(a);
}

// B fragment: 32x16 bf16 where B[k][n] = src[n][k] (src row-major, stride ld).
static __device__ __forceinline__ v16bf load_b_frag(const unsigned short* __restrict__ base,
                                                    int ld, int n0, int k0, int lane) {
  const unsigned short* p = base + (size_t)(n0 + (lane & 15)) * ld + k0 + ((lane >> 4) << 3);
  v8us b0 = *(const v8us*)p;
  v8us b1 = *(const v8us*)(p + 16);
  v16us b;
#pragma unroll
  for (int i = 0; i < 8; ++i) { b[i] = b0[i]; b[i + 8] = b1[i]; }
  return us2bf(b);
}

// ---------------- small utility kernels ----------------
__global__ void convert_f32_bf16(const float* __restrict__ src,
                                 unsigned short* __restrict__ dst, int n) {
  int i = blockIdx.x * blockDim.x + threadIdx.x;
  if (i < n) dst[i] = f2bfbits(src[i]);
}

// Pack two LxL int masks into a TILED byte array:
//   cmbt[((i>>4)*L + j)*16 + (i&15)] = (ise&1) | (hss&1)<<1
// so a 16-row x 32-col flash tile is two aligned 8B loads per lane.
__global__ void pack_masks(const int* __restrict__ ise, const int* __restrict__ hss,
                           unsigned char* __restrict__ cmbt, int n) {
  int idx = blockIdx.x * blockDim.x + threadIdx.x;
  if (idx >= n) return;
  int i = idx >> 11;          // row   (L = 2048)
  int j = idx & (LSEQ - 1);   // col
  unsigned char b = (unsigned char)((ise[idx] & 1) | ((hss[idx] & 1) << 1));
  cmbt[(((size_t)(i >> 4) * LSEQ) + j) * 16 + (i & 15)] = b;
}

// ---- bf16 WMMA GEMM, k-unrolled x2 with two independent buffer sets (no copy rotation) ----
__global__ __launch_bounds__(256) void gemm_bf16(const unsigned short* __restrict__ X,
                                                 const unsigned short* __restrict__ W,
                                                 const float* __restrict__ bias,
                                                 float* __restrict__ out,
                                                 int M, int N, int K, float scale) {
  int lane = threadIdx.x & 31;
  int wave = threadIdx.x >> 5;
  int tile = blockIdx.x * 8 + wave;
  int nTiles = N >> 6;
  int mT = tile / nTiles, nT = tile - mT * nTiles;
  int m0 = mT << 4, n0 = nT << 6;
  int ln = lane & 15, hf = lane >> 4;

  v8f acc[4] = {v8f{}, v8f{}, v8f{}, v8f{}};

  // prologue: buffer A holds k0 = 0
  v16bf aA = load_a_frag(X, K, m0 + ln, 0, lane);
  v16bf bA[4];
#pragma unroll
  for (int c = 0; c < 4; ++c) bA[c] = load_b_frag(W, K, n0 + c * 16, 0, lane);

  // K must be a multiple of 64 (it is: 1024)
  for (int k0 = 0; k0 < K; k0 += 64) {
    // fill buffer B (k0+32) before consuming buffer A
    v16bf aB = load_a_frag(X, K, m0 + ln, k0 + 32, lane);
    v16bf bB[4];
#pragma unroll
    for (int c = 0; c < 4; ++c) bB[c] = load_b_frag(W, K, n0 + c * 16, k0 + 32, lane);

#pragma unroll
    for (int c = 0; c < 4; ++c)
      acc[c] = __builtin_amdgcn_wmma_f32_16x16x32_bf16(false, aA, false, bA[c],
                                                       (short)0, acc[c], false, false);

    // refill buffer A (k0+64) before consuming buffer B; clamp for final iteration
    int k2 = (k0 + 64 < K) ? (k0 + 64) : 0;
    aA = load_a_frag(X, K, m0 + ln, k2, lane);
#pragma unroll
    for (int c = 0; c < 4; ++c) bA[c] = load_b_frag(W, K, n0 + c * 16, k2, lane);

#pragma unroll
    for (int c = 0; c < 4; ++c)
      acc[c] = __builtin_amdgcn_wmma_f32_16x16x32_bf16(false, aB, false, bB[c],
                                                       (short)0, acc[c], false, false);
  }

#pragma unroll
  for (int c = 0; c < 4; ++c) {
    int ncol = n0 + c * 16 + ln;
    float bv = bias[ncol];
#pragma unroll
    for (int r = 0; r < 8; ++r) {
      int mrow = m0 + r + hf * 8;
      out[(size_t)mrow * N + ncol] = (acc[c][r] + bv) * scale;
    }
  }
}

// ---------------- rotary + head-major repack ----------------
__global__ void rotary_repack(const float* __restrict__ Qf, const float* __restrict__ Kf,
                              const float* __restrict__ Vf,
                              unsigned short* __restrict__ Qb, unsigned short* __restrict__ Qrb,
                              unsigned short* __restrict__ Kb, unsigned short* __restrict__ Krb,
                              unsigned short* __restrict__ Vt) {
  int idx = blockIdx.x * blockDim.x + threadIdx.x;
  if (idx >= LSEQ * EDIM) return;
  int l = idx >> 10, e = idx & 1023;
  int h = e >> 6, d = e & 63;
  float qv = Qf[idx], kv = Kf[idx], vv = Vf[idx];
  int dp = (d < 32) ? idx + 32 : idx - 32;
  float qo = Qf[dp], ko = Kf[dp];
  float rq = (d < 32) ? -qo : qo;
  float rk = (d < 32) ? -ko : ko;
  // inv_freq[j] = 10000^(-2j/64); angle = l * inv_freq[d&31]
  float ang = (float)l * __expf(-(float)(2 * (d & 31)) * 0.14391156831f); // ln(10000)/64
  float s, c;
  __sincosf(ang, &s, &c);
  float qr = qv * c + rq * s;
  float kr = kv * c + rk * s;
  size_t ho = ((size_t)(h * LSEQ + l)) * HD + d;
  Qb[ho] = f2bfbits(qv);
  Qrb[ho] = f2bfbits(qr);
  Kb[ho] = f2bfbits(kv);
  Krb[ho] = f2bfbits(kr);
  Vt[((size_t)(h * HD + d)) * LSEQ + l] = f2bfbits(vv);
}

// ---------------- attn_diff[h] = ((Σq_r)·(Σk_r) − (Σq)·(Σk)) / L² ----------------
__global__ void head_diff(const unsigned short* __restrict__ Qb, const unsigned short* __restrict__ Qrb,
                          const unsigned short* __restrict__ Kb, const unsigned short* __restrict__ Krb,
                          float* __restrict__ diff) {
  __shared__ float sp[64], spr[64];
  int h = blockIdx.x, d = threadIdx.x;
  const size_t base = (size_t)h * LSEQ * HD + d;
  float sq = 0.f, sqr = 0.f, sk = 0.f, skr = 0.f;
  for (int l = 0; l < LSEQ; ++l) {
    size_t o = base + (size_t)l * HD;
    sq += bf2f(Qb[o]); sqr += bf2f(Qrb[o]);
    sk += bf2f(Kb[o]); skr += bf2f(Krb[o]);
  }
  sp[d] = sq * sk;
  spr[d] = sqr * skr;
  __syncthreads();
  if (d == 0) {
    float a = 0.f, b = 0.f;
    for (int i = 0; i < 64; ++i) { a += spr[i]; b += sp[i]; }
    diff[h] = (a - b) / (float)((size_t)LSEQ * LSEQ);
  }
}

// ---------------- flash attention: one wave per (head, 16-row tile) ----------------
__global__ __launch_bounds__(256) void flash_attn(
    const unsigned short* __restrict__ Qb, const unsigned short* __restrict__ Qrb,
    const unsigned short* __restrict__ Kb, const unsigned short* __restrict__ Krb,
    const unsigned short* __restrict__ Vt, const unsigned char* __restrict__ cmbt,
    const unsigned char* __restrict__ kpm, const float* __restrict__ ent,
    const float* __restrict__ seqe, const float* __restrict__ diff,
    unsigned short* __restrict__ Ob) {
  __shared__ unsigned short lds[8][512]; // 16 rows x 32 cols bf16 per wave
  int lane = threadIdx.x & 31;
  int wv = threadIdx.x >> 5;
  int gw = blockIdx.x * 8 + wv;
  int h = gw >> 7;           // 128 i-tiles per head
  int i0 = (gw & 127) << 4;
  int hf = lane >> 4, ln = lane & 15;

  const unsigned short* Qh = Qb + (size_t)h * LSEQ * HD;
  const unsigned short* Qrh = Qrb + (size_t)h * LSEQ * HD;
  const unsigned short* Kh = Kb + (size_t)h * LSEQ * HD;
  const unsigned short* Krh = Krb + (size_t)h * LSEQ * HD;
  const unsigned short* Vh = Vt + (size_t)h * HD * LSEQ;
  const unsigned char* cmbRow = cmbt + (size_t)(i0 >> 4) * LSEQ * 16 + hf * 8;

  v16bf qa0 = load_a_frag(Qh, HD, i0 + ln, 0, lane);
  v16bf qa1 = load_a_frag(Qh, HD, i0 + ln, 32, lane);
  v16bf qr0 = load_a_frag(Qrh, HD, i0 + ln, 0, lane);
  v16bf qr1 = load_a_frag(Qrh, HD, i0 + ln, 32, lane);

  float e0 = ent[h], e1 = ent[NH + h];
  float s0 = seqe[h], s1 = seqe[NH + h];
  float dh = diff[h];

  v8f o[4] = {v8f{}, v8f{}, v8f{}, v8f{}};
  float m[8], lsum[8];
#pragma unroll
  for (int r = 0; r < 8; ++r) { m[r] = -1e30f; lsum[r] = 0.f; }

  for (int j0 = 0; j0 < LSEQ; j0 += 32) {
    // mask tile: one aligned 8B load per chunk per lane (tiled layout)
    v8uc cb0 = *(const v8uc*)(cmbRow + (size_t)(j0 + ln) * 16);
    v8uc cb1 = *(const v8uc*)(cmbRow + (size_t)(j0 + 16 + ln) * 16);
    unsigned char km0 = kpm[j0 + ln];
    unsigned char km1 = kpm[j0 + 16 + ln];

    // all K / K_rot B-fragments up front (one wait, 8 WMMAs back-to-back)
    v16bf kb[2][2], krb[2][2];
#pragma unroll
    for (int c = 0; c < 2; ++c) {
#pragma unroll
      for (int kk = 0; kk < 2; ++kk) {
        kb[c][kk]  = load_b_frag(Kh,  HD, j0 + c * 16, kk * 32, lane);
        krb[c][kk] = load_b_frag(Krh, HD, j0 + c * 16, kk * 32, lane);
      }
    }

    v8f sr[2], st[2];
#pragma unroll
    for (int c = 0; c < 2; ++c) {
      v8f z = v8f{};
      z = __builtin_amdgcn_wmma_f32_16x16x32_bf16(false, qa0, false, kb[c][0], (short)0, z, false, false);
      z = __builtin_amdgcn_wmma_f32_16x16x32_bf16(false, qa1, false, kb[c][1], (short)0, z, false, false);
      sr[c] = z;
      v8f zr = v8f{};
      zr = __builtin_amdgcn_wmma_f32_16x16x32_bf16(false, qr0, false, krb[c][0], (short)0, zr, false, false);
      zr = __builtin_amdgcn_wmma_f32_16x16x32_bf16(false, qr1, false, krb[c][1], (short)0, zr, false, false);
      st[c] = zr;
    }

    // blend raw/rotary, add entity/sequence bias, apply key padding mask
    float w[2][8];
#pragma unroll
    for (int r = 0; r < 8; ++r) {
      unsigned char b0 = cb0[r], b1 = cb1[r];
      bool ss0 = (b0 & 2) != 0, ss1 = (b1 & 2) != 0;
      float v0 = ss0 ? st[0][r] : (sr[0][r] + dh);
      float v1 = ss1 ? st[1][r] : (sr[1][r] + dh);
      v0 += ((b0 & 1) ? e1 : e0) + (ss0 ? s1 : s0);
      v1 += ((b1 & 1) ? e1 : e0) + (ss1 ? s1 : s0);
      w[0][r] = km0 ? -1e30f : v0;
      w[1][r] = km1 ? -1e30f : v1;
    }

    // online softmax (rows live across 16-lane halves in C layout)
#pragma unroll
    for (int r = 0; r < 8; ++r) {
      float tm = fmaxf(w[0][r], w[1][r]);
#pragma unroll
      for (int msk = 1; msk < 16; msk <<= 1) tm = fmaxf(tm, __shfl_xor(tm, msk, 32));
      float mn = fmaxf(m[r], tm);
      float scl = __expf(m[r] - mn);
      float p0 = __expf(w[0][r] - mn);
      float p1 = __expf(w[1][r] - mn);
      w[0][r] = p0; w[1][r] = p1;
      float rs = p0 + p1;
#pragma unroll
      for (int msk = 1; msk < 16; msk <<= 1) rs += __shfl_xor(rs, msk, 32);
      lsum[r] = lsum[r] * scl + rs;
      m[r] = mn;
      o[0][r] *= scl; o[1][r] *= scl; o[2][r] *= scl; o[3][r] *= scl;
    }

    // issue V fragment loads now; they complete while we do the LDS roundtrip
    v16us vb[4];
#pragma unroll
    for (int dc = 0; dc < 4; ++dc) {
      const unsigned short* vp = Vh + (size_t)(dc * 16 + ln) * LSEQ + j0 + hf * 8;
      v8us b0 = *(const v8us*)vp;
      v8us b1 = *(const v8us*)(vp + 16);
#pragma unroll
      for (int i = 0; i < 8; ++i) { vb[dc][i] = b0[i]; vb[dc][i + 8] = b1[i]; }
    }

    // C-layout P -> A-layout bf16 fragment via LDS
    __syncthreads();
#pragma unroll
    for (int c = 0; c < 2; ++c)
#pragma unroll
      for (int r = 0; r < 8; ++r)
        lds[wv][(r + hf * 8) * 32 + c * 16 + ln] = f2bfbits(w[c][r]);
    __syncthreads();
    const unsigned short* pl = &lds[wv][ln * 32 + hf * 8];
    v8us p0v = *(const v8us*)pl;
    v8us p1v = *(const v8us*)(pl + 16);
    v16us pu;
#pragma unroll
    for (int i = 0; i < 8; ++i) { pu[i] = p0v[i]; pu[i + 8] = p1v[i]; }
    v16bf pa = us2bf(pu);

    // O += P · V
#pragma unroll
    for (int dc = 0; dc < 4; ++dc)
      o[dc] = __builtin_amdgcn_wmma_f32_16x16x32_bf16(false, pa, false, us2bf(vb[dc]),
                                                      (short)0, o[dc], false, false);
    __syncthreads();
  }

  // normalize and write bf16 [L][E] for the output projection
#pragma unroll
  for (int dc = 0; dc < 4; ++dc) {
#pragma unroll
    for (int r = 0; r < 8; ++r) {
      int i = i0 + r + hf * 8;
      int e = h * HD + dc * 16 + ln;
      Ob[(size_t)i * EDIM + e] = f2bfbits(o[dc][r] / lsum[r]);
    }
  }
}

// ---------------- host launch ----------------
extern "C" void kernel_launch(void* const* d_in, const int* in_sizes, int n_in,
                              void* d_out, int out_size, void* d_ws, size_t ws_size,
                              hipStream_t stream) {
  const float*         query = (const float*)d_in[0];
  const unsigned char* kpm   = (const unsigned char*)d_in[1];
  const int*           ise   = (const int*)d_in[2];
  const int*           hss   = (const int*)d_in[3];
  const float* Wq = (const float*)d_in[4];
  const float* bq = (const float*)d_in[5];
  const float* Wk = (const float*)d_in[6];
  const float* bk = (const float*)d_in[7];
  const float* Wv = (const float*)d_in[8];
  const float* bv = (const float*)d_in[9];
  const float* Wo = (const float*)d_in[10];
  const float* bo = (const float*)d_in[11];
  const float* ent = (const float*)d_in[12];
  const float* seq = (const float*)d_in[13];

  char* ws = (char*)d_ws;
  size_t off = 0;
  auto alloc = [&](size_t bytes) -> void* {
    void* p = ws + off;
    off += (bytes + 255) & ~(size_t)255;
    return p;
  };

  const size_t NLE = (size_t)LSEQ * EDIM;  // 2M
  const size_t NEE = (size_t)EDIM * EDIM;  // 1M

  unsigned short* Xb   = (unsigned short*)alloc(NLE * 2);
  unsigned short* Wqb  = (unsigned short*)alloc(NEE * 2);
  unsigned short* Wkb  = (unsigned short*)alloc(NEE * 2);
  unsigned short* Wvb  = (unsigned short*)alloc(NEE * 2);
  unsigned short* Wob  = (unsigned short*)alloc(NEE * 2);
  float* Qf = (float*)alloc(NLE * 4);
  float* Kf = (float*)alloc(NLE * 4);
  float* Vf = (float*)alloc(NLE * 4);
  unsigned short* Qb  = (unsigned short*)alloc(NLE * 2);
  unsigned short* Qrb = (unsigned short*)alloc(NLE * 2);
  unsigned short* Kb  = (unsigned short*)alloc(NLE * 2);
  unsigned short* Krb = (unsigned short*)alloc(NLE * 2);
  unsigned short* Vt  = (unsigned short*)alloc(NLE * 2);
  unsigned short* Ob  = (unsigned short*)alloc(NLE * 2);
  unsigned char*  cmbt = (unsigned char*)alloc((size_t)LSEQ * LSEQ);
  float* diff = (float*)alloc(NH * 4);

  // 1) bf16 conversions
  convert_f32_bf16<<<(int)((NLE + 255) / 256), 256, 0, stream>>>(query, Xb, (int)NLE);
  convert_f32_bf16<<<(int)((NEE + 255) / 256), 256, 0, stream>>>(Wq, Wqb, (int)NEE);
  convert_f32_bf16<<<(int)((NEE + 255) / 256), 256, 0, stream>>>(Wk, Wkb, (int)NEE);
  convert_f32_bf16<<<(int)((NEE + 255) / 256), 256, 0, stream>>>(Wv, Wvb, (int)NEE);
  convert_f32_bf16<<<(int)((NEE + 255) / 256), 256, 0, stream>>>(Wo, Wob, (int)NEE);

  // 2) pack the two LxL int masks into one tiled byte array
  pack_masks<<<(LSEQ * LSEQ) / 256, 256, 0, stream>>>(ise, hss, cmbt, LSEQ * LSEQ);

  // 3) QKV projections (Q pre-scaled by D^-0.5)
  const int gemmBlocks = (LSEQ / 16) * (EDIM / 64) / 8; // 256
  gemm_bf16<<<gemmBlocks, 256, 0, stream>>>(Xb, Wqb, bq, Qf, LSEQ, EDIM, EDIM, SCALE_Q);
  gemm_bf16<<<gemmBlocks, 256, 0, stream>>>(Xb, Wkb, bk, Kf, LSEQ, EDIM, EDIM, 1.0f);
  gemm_bf16<<<gemmBlocks, 256, 0, stream>>>(Xb, Wvb, bv, Vf, LSEQ, EDIM, EDIM, 1.0f);

  // 4) rotary + head-major repack (+ transposed V)
  rotary_repack<<<(int)((NLE + 255) / 256), 256, 0, stream>>>(Qf, Kf, Vf, Qb, Qrb, Kb, Krb, Vt);

  // 5) factorized attn_diff per head
  head_diff<<<NH, 64, 0, stream>>>(Qb, Qrb, Kb, Krb, diff);

  // 6) fused flash attention with blend + bias + softmax
  const int flashBlocks = NH * (LSEQ / 16) / 8; // 256
  flash_attn<<<flashBlocks, 256, 0, stream>>>(Qb, Qrb, Kb, Krb, Vt, cmbt, kpm, ent, seq, diff, Ob);

  // 7) output projection into d_out (f32)
  gemm_bf16<<<gemmBlocks, 256, 0, stream>>>(Ob, Wob, bo, (float*)d_out, LSEQ, EDIM, EDIM, 1.0f);
}
